// LAGCNII_77129022701605
// MI455X (gfx1250) — compile-verified, matrix operand order checked by reading, not run on hardware
//
#include <hip/hip_runtime.h>
#include <cmath>

#define D_IN  256
#define D_HID 128
#define CH    256
#define CLS   40
#define NLAYER 8
#define LSTR  20   // LDS row stride in words: 80B rows -> 16B-aligned frag quads

typedef __attribute__((ext_vector_type(16))) _Float16 v16h;
typedef __attribute__((ext_vector_type(8)))  float    v8f;
typedef __attribute__((ext_vector_type(2)))  unsigned v2u;
typedef __attribute__((ext_vector_type(4)))  unsigned v4u;

__device__ __forceinline__ unsigned pack2(float a, float b) {
  union { decltype(__builtin_amdgcn_cvt_pkrtz(0.f, 0.f)) h; unsigned u; } r;
  r.h = __builtin_amdgcn_cvt_pkrtz(a, b);   // one v_cvt_pk_rtz_f16_f32
  return r.u;
}

// Fragment layout per CDNA5 ISA (16-bit A 16x32).  LDS tile = [row][LSTR] u32
// words, word j = f16 pair (k=2j,2j+1) of a 32-wide K slab.
// lane = half*16 + m ; VGPR j   -> K = half*8 + 2j   -> words h4..h4+3
//                      VGPR 4+j -> K = 16 + half*8+2j -> words 8+h4..8+h4+3
// Both quads are 16B aligned (80*row + 16*half): 2x ds_load_b128 per frag.
__device__ __forceinline__ v16h load_frag(const unsigned* lds, int rowBase, int lane) {
  const unsigned* p = lds + (rowBase + (lane & 15)) * LSTR + (lane >> 4) * 4;
  union { v4u q[2]; v16h v; } r;
  r.q[0] = *(const v4u*)(p);
  r.q[1] = *(const v4u*)(p + 8);
  return r.v;
}

#define WMMA_F16(A, B, C) \
  __builtin_amdgcn_wmma_f32_16x16x32_f16(false, (A), false, (B), (short)0, (C), false, false)

// ---------------------------------------------------------------------------
// Shared GEMM mainloop: block tile 128(M) x 64(N), K=256, 8 waves as 4x2,
// each wave a 32x32 region (2x2 WMMA tiles).
//  - Whole 64x256 W tile (32KB) loaded to LDS once in the prologue
//    -> no per-iteration W traffic, no sinkable W staging.
//  - A tile double-buffered: slab k+1's b128 global loads issue before the
//    frag reads + WMMAs of slab k; cvt+ds_store afterwards, so the loadcnt
//    wait lands after the matrix work.  One barrier per K-step.
// A row stride in memory is always 256 floats (D_IN == CH == 256).
// RESIDUAL: A = 0.9*A0 + 0.1*A1, else A = A0.
// ---------------------------------------------------------------------------
template <bool RESIDUAL>
__device__ __forceinline__ void gemm_core(
    const float* __restrict__ A0, const float* __restrict__ A1,
    const unsigned* __restrict__ wtp, int mBase, int nBase, int nNodes,
    unsigned* __restrict__ lds_x,   // [2][128*LSTR]
    unsigned* __restrict__ lds_w,   // [8][64*LSTR]
    v8f acc[2][2]) {
  const int tid  = threadIdx.x;
  const int lane = tid & 31;
  const int wave = tid >> 5;
  const int mWave = (wave & 3) * 32;
  const int nWave = (wave >> 2) * 32;
  const int q  = tid & 7;    // float4 slot within A row (k = ks + 4q)
  const int m0 = tid >> 3;   // A rows m0 + 32i (i<4)

  float4 ra[4], rb[4];

  auto load_a = [&](int ks) {
#pragma unroll
    for (int i = 0; i < 4; ++i) {
      int gm  = mBase + m0 + 32 * i;
      int gmc = gm < nNodes ? gm : nNodes - 1;          // clamp, no exec branch
      size_t off = (size_t)gmc * 256 + ks + q * 4;
      ra[i] = *(const float4*)(A0 + off);
      if (RESIDUAL) rb[i] = *(const float4*)(A1 + off);
    }
  };

  auto store_a = [&](unsigned* bx) {
#pragma unroll
    for (int i = 0; i < 4; ++i) {
      int m = m0 + 32 * i;
      float s = (mBase + m < nNodes) ? 1.f : 0.f;       // zero OOB rows
      float vx, vy, vz, vw;
      if (RESIDUAL) {
        vx = (0.9f * ra[i].x + 0.1f * rb[i].x) * s;
        vy = (0.9f * ra[i].y + 0.1f * rb[i].y) * s;
        vz = (0.9f * ra[i].z + 0.1f * rb[i].z) * s;
        vw = (0.9f * ra[i].w + 0.1f * rb[i].w) * s;
      } else {
        vx = ra[i].x * s; vy = ra[i].y * s; vz = ra[i].z * s; vw = ra[i].w * s;
      }
      v2u t = {pack2(vx, vy), pack2(vz, vw)};
      *(v2u*)(bx + m * LSTR + 2 * q) = t;               // 8B-aligned ds_store_b64
    }
  };

  // ---- prologue: full W tile (64 rows x 256 k) + A slab 0 -----------------
  v4u wreg[8];
#pragma unroll
  for (int i = 0; i < 8; ++i) {
    int flat = tid + 256 * i;          // 2048 v4u = 64 rows x 32 quads
    int n = flat >> 5, c4 = flat & 31;
    wreg[i] = *(const v4u*)(wtp + (size_t)(nBase + n) * 128 + c4 * 4);
  }
  load_a(0);
#pragma unroll
  for (int i = 0; i < 8; ++i) {
    int flat = tid + 256 * i;
    int n = flat >> 5, c4 = flat & 31;
    int slab = c4 >> 2, w4 = (c4 & 3) * 4;
    *(v4u*)(lds_w + slab * (64 * LSTR) + n * LSTR + w4) = wreg[i];  // b128
  }
  store_a(lds_x);
  __syncthreads();

  // ---- mainloop -----------------------------------------------------------
#pragma unroll
  for (int ksi = 0; ksi < 8; ++ksi) {
    unsigned* bx = lds_x + (ksi & 1) * (128 * LSTR);
    const unsigned* bw = lds_w + ksi * (64 * LSTR);
    if (ksi < 7) load_a((ksi + 1) * 32);      // issue prefetch loads
    v16h a0 = load_frag(bx, mWave,      lane);
    v16h a1 = load_frag(bx, mWave + 16, lane);
    v16h b0 = load_frag(bw, nWave,      lane);
    v16h b1 = load_frag(bw, nWave + 16, lane);
    acc[0][0] = WMMA_F16(a0, b0, acc[0][0]);
    acc[0][1] = WMMA_F16(a0, b1, acc[0][1]);
    acc[1][0] = WMMA_F16(a1, b0, acc[1][0]);
    acc[1][1] = WMMA_F16(a1, b1, acc[1][1]);
    if (ksi < 7) {
      store_a(lds_x + ((ksi & 1) ^ 1) * (128 * LSTR));  // loadcnt wait lands here
      __syncthreads();
    }
  }
}

// ---------------------------------------------------------------------------
// Weight prep: f32 -> f16, transposed so B-frag loads mirror A-frag loads.
// gcnWT[l][n][k], linWT[v][n][k], outWT[n][k] (n padded to 64, zero filled)
// ---------------------------------------------------------------------------
__global__ void prep_weights(const float* __restrict__ gcn_w,
                             const float* __restrict__ lin_w,
                             const float* __restrict__ out_w,
                             _Float16* __restrict__ gcnWT,
                             _Float16* __restrict__ linWT,
                             _Float16* __restrict__ outWT) {
  int idx = blockIdx.x * blockDim.x + threadIdx.x;
  const int GCN = NLAYER * CH * CH;
  const int LIN = 2 * D_HID * D_IN;
  const int OUT = 64 * CH;
  if (idx < GCN) {
    int l = idx >> 16, r = idx & 65535, n = r >> 8, k = r & 255;
    gcnWT[idx] = (_Float16)gcn_w[l * 65536 + k * 256 + n];
  } else if (idx < GCN + LIN) {
    int jj = idx - GCN;
    int v = jj >> 15, r = jj & 32767, n = r >> 8, k = r & 255;
    linWT[jj] = (_Float16)lin_w[v * (D_IN * D_HID) + k * D_HID + n];
  } else if (idx < GCN + LIN + OUT) {
    int jj = idx - GCN - LIN;
    int n = jj >> 8, k = jj & 255;
    outWT[jj] = (n < CLS) ? (_Float16)out_w[k * CLS + n] : (_Float16)0.0f;
  }
}

// ---------------------------------------------------------------------------
// Edge scatter: agg[dst] += h[src]  (fp32 atomics; h/agg are L2-resident)
// ---------------------------------------------------------------------------
__global__ __launch_bounds__(256)
void scatter_add(const int* __restrict__ src, const int* __restrict__ dst,
                 const float* __restrict__ h, float* __restrict__ agg, int nEdges) {
  int idx = blockIdx.x * blockDim.x + threadIdx.x;
  int e = idx >> 6;
  if (e >= nEdges) return;
  int c = (idx & 63) * 4;
  int s = src[e], d = dst[e];
  float4 v = *(const float4*)(h + (size_t)s * CH + c);
  float* ap = agg + (size_t)d * CH + c;
  atomicAdd(ap + 0, v.x);
  atomicAdd(ap + 1, v.y);
  atomicAdd(ap + 2, v.z);
  atomicAdd(ap + 3, v.w);
}

// ---------------------------------------------------------------------------
// Input linears: h0 = concat(relu(x0@W0+b0), relu(x1@W1+b1)); also copied to h
// grid: (ceil(N/128), 2, 2 views)
// ---------------------------------------------------------------------------
__global__ __launch_bounds__(256)
void lin_gemm(const float* __restrict__ x0, const float* __restrict__ x1,
              const _Float16* __restrict__ linWT, const float* __restrict__ linB,
              float* __restrict__ h0, float* __restrict__ h, int nNodes) {
  __shared__ unsigned lds_x[2 * 128 * LSTR];
  __shared__ unsigned lds_w[8 * 64 * LSTR];
  const int lane = threadIdx.x & 31;
  const int wave = threadIdx.x >> 5;
  const int view  = blockIdx.z;
  const int mBase = blockIdx.x * 128;
  const int nBase = blockIdx.y * 64;
  const int mWave = (wave & 3) * 32;
  const int nWave = (wave >> 2) * 32;
  const float* x = view ? x1 : x0;
  v8f acc[2][2] = {};
  gemm_core<false>(x, nullptr,
                   (const unsigned*)(linWT + (size_t)view * D_HID * D_IN),
                   mBase, nBase, nNodes, lds_x, lds_w, acc);
  auto epi = [&](bool guard) {
#pragma unroll
    for (int ti = 0; ti < 2; ++ti)
#pragma unroll
      for (int tj = 0; tj < 2; ++tj) {
        int r0 = mBase + mWave + ti * 16 + (lane >> 4) * 8;
        int cl = nBase + nWave + tj * 16 + (lane & 15);   // local col 0..127
        int gc = view * D_HID + cl;                        // global col 0..255
        float b = linB[view * D_HID + cl];
#pragma unroll
        for (int v = 0; v < 8; ++v) {
          int row = r0 + v;
          if (!guard || row < nNodes) {
            float o = acc[ti][tj][v] + b;
            o = o > 0.f ? o : 0.f;
            h0[(size_t)row * CH + gc] = o;
            h [(size_t)row * CH + gc] = o;
          }
        }
      }
  };
  if (mBase + 128 <= nNodes) epi(false); else epi(true);
}

// ---------------------------------------------------------------------------
// GCNII layer: x = 0.9*agg + 0.1*h0 ; h = relu((1-beta)*x + beta*(x @ W))
// grid: (ceil(N/128), 4)
// ---------------------------------------------------------------------------
__global__ __launch_bounds__(256)
void layer_gemm(const float* __restrict__ agg, const float* __restrict__ h0,
                float* __restrict__ h, const _Float16* __restrict__ wT,
                float beta, int nNodes) {
  __shared__ unsigned lds_x[2 * 128 * LSTR];
  __shared__ unsigned lds_w[8 * 64 * LSTR];
  const int lane = threadIdx.x & 31;
  const int wave = threadIdx.x >> 5;
  const int mBase = blockIdx.x * 128;
  const int nBase = blockIdx.y * 64;
  const int mWave = (wave & 3) * 32;
  const int nWave = (wave >> 2) * 32;
  v8f acc[2][2] = {};
  gemm_core<true>(agg, h0, (const unsigned*)wT, mBase, nBase, nNodes,
                  lds_x, lds_w, acc);
  const float omb = 1.0f - beta;
  auto epi = [&](bool guard) {
#pragma unroll
    for (int ti = 0; ti < 2; ++ti)
#pragma unroll
      for (int tj = 0; tj < 2; ++tj) {
        int r0 = mBase + mWave + ti * 16 + (lane >> 4) * 8;
        int c  = nBase + nWave + tj * 16 + (lane & 15);
#pragma unroll
        for (int v = 0; v < 8; ++v) {
          int row = r0 + v;
          if (!guard || row < nNodes) {
            size_t off = (size_t)row * CH + c;
            float x = 0.9f * agg[off] + 0.1f * h0[off];   // fp32 identity path
            float o = omb * x + beta * acc[ti][tj][v];
            h[off] = o > 0.f ? o : 0.f;
          }
        }
      }
  };
  if (mBase + 128 <= nNodes) epi(false); else epi(true);
}

// ---------------------------------------------------------------------------
// Output head: out = h @ out_w + out_b   (cols 0..39 valid, W^T padded to 64)
// grid: (ceil(N/128), 1)
// ---------------------------------------------------------------------------
__global__ __launch_bounds__(256)
void out_gemm(const float* __restrict__ h, const _Float16* __restrict__ outWT,
              const float* __restrict__ outB, float* __restrict__ out, int nNodes) {
  __shared__ unsigned lds_x[2 * 128 * LSTR];
  __shared__ unsigned lds_w[8 * 64 * LSTR];
  const int lane = threadIdx.x & 31;
  const int wave = threadIdx.x >> 5;
  const int mBase = blockIdx.x * 128;
  const int mWave = (wave & 3) * 32;
  const int nWave = (wave >> 2) * 32;
  v8f acc[2][2] = {};
  gemm_core<false>(h, nullptr, (const unsigned*)outWT, mBase, 0, nNodes,
                   lds_x, lds_w, acc);
  auto epi = [&](bool guard) {
#pragma unroll
    for (int ti = 0; ti < 2; ++ti)
#pragma unroll
      for (int tj = 0; tj < 2; ++tj) {
        int r0 = mBase + mWave + ti * 16 + (lane >> 4) * 8;
        int c  = nWave + tj * 16 + (lane & 15);
#pragma unroll
        for (int v = 0; v < 8; ++v) {
          int row = r0 + v;
          if ((!guard || row < nNodes) && c < CLS)
            out[(size_t)row * CLS + c] = acc[ti][tj][v] + outB[c];
        }
      }
  };
  if (mBase + 128 <= nNodes) epi(false); else epi(true);
}

// ---------------------------------------------------------------------------
extern "C" void kernel_launch(void* const* d_in, const int* in_sizes, int n_in,
                              void* d_out, int out_size, void* d_ws, size_t ws_size,
                              hipStream_t stream) {
  (void)n_in; (void)out_size; (void)ws_size;
  const float* x0    = (const float*)d_in[0];
  const float* x1    = (const float*)d_in[1];
  const int*   ei    = (const int*)d_in[2];
  const float* lin_w = (const float*)d_in[3];
  const float* lin_b = (const float*)d_in[4];
  const float* gcn_w = (const float*)d_in[5];
  const float* out_w = (const float*)d_in[6];
  const float* out_b = (const float*)d_in[7];

  const int nNodes = in_sizes[0] / D_IN;
  const int nEdges = in_sizes[2] / 2;

  char* ws = (char*)d_ws;
  const size_t hBytes = (size_t)nNodes * CH * sizeof(float);
  float* h0  = (float*)(ws);
  float* h   = (float*)(ws + hBytes);
  float* agg = (float*)(ws + 2 * hBytes);
  char*  wts = ws + 3 * hBytes;
  _Float16* gcnWT = (_Float16*)wts;
  _Float16* linWT = (_Float16*)(wts + sizeof(_Float16) * (size_t)NLAYER * CH * CH);
  _Float16* outWT = (_Float16*)(wts + sizeof(_Float16) * ((size_t)NLAYER * CH * CH + 2 * D_HID * D_IN));

  {
    int total = NLAYER * CH * CH + 2 * D_HID * D_IN + 64 * CH;
    prep_weights<<<(total + 255) / 256, 256, 0, stream>>>(gcn_w, lin_w, out_w,
                                                          gcnWT, linWT, outWT);
  }

  const int mTiles = (nNodes + 127) / 128;

  lin_gemm<<<dim3(mTiles, 2, 2), 256, 0, stream>>>(x0, x1, linWT, lin_b, h0, h, nNodes);

  for (int l = 0; l < NLAYER; ++l) {
    (void)hipMemsetAsync(agg, 0, hBytes, stream);
    scatter_add<<<((size_t)nEdges * 64 + 255) / 256, 256, 0, stream>>>(
        ei, ei + nEdges, h, agg, nEdges);
    float beta = logf(0.5f / (float)(l + 1) + 1.0f);
    layer_gemm<<<dim3(mTiles, CH / 64), 256, 0, stream>>>(
        agg, h0, h, gcnWT + (size_t)l * CH * CH, beta, nNodes);
  }

  out_gemm<<<dim3(mTiles, 1), 256, 0, stream>>>(h, outWT, out_b, (float*)d_out, nNodes);
}
